// ElmanRNN_pred_75617194213993
// MI455X (gfx1250) — matrix operand order repaired and checked by reading, not compiled, not run on hardware
//
#include <hip/hip_runtime.h>
#include <hip/hip_bf16.h>
#include <math.h>

// Problem dims (from reference setup_inputs)
constexpr int B = 64, S = 512, I = 512, H = 1024, O = 512;

typedef __attribute__((ext_vector_type(16))) __bf16 v16bf;
typedef __attribute__((ext_vector_type(8)))  float  v8f;
typedef int v4i_ __attribute__((vector_size(16)));  // b128 payload type

union ABFrag { uint4 q[2]; v16bf v; };

#define WMMA_BF16(a, b, c) \
  __builtin_amdgcn_wmma_f32_16x16x32_bf16(false, (a), false, (b), (short)0, (c), false, false)

#define AS1 __attribute__((address_space(1)))
#define AS3 __attribute__((address_space(3)))

#if __has_builtin(__builtin_amdgcn_global_load_async_to_lds_b128)
#define HAVE_ASYNC_COPY 1
#else
#define HAVE_ASYNC_COPY 0
#endif

#if __has_builtin(__builtin_amdgcn_s_wait_asynccnt)
#define WAIT_ASYNC(n) __builtin_amdgcn_s_wait_asynccnt(n)
#else
#define WAIT_ASYNC(n) asm volatile("s_wait_asynccnt %0" ::"i"(n) : "memory")
#endif

__device__ __forceinline__ unsigned short f32_to_bf16_rne(float f) {
  unsigned int u = __float_as_uint(f);
  u += 0x7FFFu + ((u >> 16) & 1u);
  return (unsigned short)(u >> 16);
}
__device__ __forceinline__ float bf16_to_f32(unsigned short h) {
  return __uint_as_float(((unsigned int)h) << 16);
}

// A fragment, 16x32 bf16 (ISA 7.12.2): lane holds row M=lane&15;
// chunk0 = K[(lane>>4)*8 .. +7], chunk1 = same + 16. Two 16B loads.
// Works for global rows (stride ldk via rowp) and LDS-staged rows (stride 32).
__device__ __forceinline__ v16bf load_a_frag(const unsigned short* rowp, int kk) {
  const int lane = threadIdx.x & 31;
  const int ksel = (lane >> 4) << 3;
  const unsigned short* p = rowp + kk + ksel;
  ABFrag f;
  f.q[0] = *(const uint4*)(p);
  f.q[1] = *(const uint4*)(p + 16);
  return f.v;
}

// B fragment, 32x16 bf16: lane holds column N=n0+(lane&15),
// chunk0 = K[kk + (lane>>4)*8 .. +7], chunk1 = same + 16. Two 16B loads.
__device__ __forceinline__ v16bf load_b_frag(const unsigned short* W, int n0, int ldk, int kk) {
  const int lane = threadIdx.x & 31;
  const int n = n0 + (lane & 15);
  const int koff = (lane >> 4) << 3;
  const unsigned short* p = W + (size_t)n * ldk + kk + koff;
  ABFrag f;
  f.q[0] = *(const uint4*)(p);
  f.q[1] = *(const uint4*)(p + 16);
  return f.v;
}

// ---------------- utility kernels ----------------
__global__ void k_f32_to_bf16(const float* __restrict__ src,
                              unsigned short* __restrict__ dst, long n) {
  long i = (long)blockIdx.x * blockDim.x + threadIdx.x;
  long stride = (long)gridDim.x * blockDim.x;
  for (; i < n; i += stride) dst[i] = f32_to_bf16_rne(src[i]);
}

__global__ void k_bf16_to_f32(const unsigned short* __restrict__ src,
                              float* __restrict__ dst, long n) {
  long i = (long)blockIdx.x * blockDim.x + threadIdx.x;
  long stride = (long)gridDim.x * blockDim.x;
  for (; i < n; i += stride) dst[i] = bf16_to_f32(src[i]);
}

__global__ void k_zero_z0(unsigned short* __restrict__ z) {
  int i = blockIdx.x * blockDim.x + threadIdx.x;  // B*H threads
  if (i < B * H) {
    int b = i >> 10;  // /H
    int h = i & (H - 1);
    z[(size_t)b * S * H + h] = 0;
  }
}

// ---------------- P = x@Wi^T + bi + bh for all t in [0,S-1) ----------------
// M rows r = t*B + b ; grid (M/16, H/64); 128 threads = 4 waves, 1 tile/wave
__global__ void k_pgemm(const unsigned short* __restrict__ xb,
                        const unsigned short* __restrict__ Wib,
                        const float* __restrict__ bi, const float* __restrict__ bh,
                        unsigned short* __restrict__ Pb) {
  const int lane = threadIdx.x & 31;
  const int wave = threadIdx.x >> 5;
  const int m0 = blockIdx.x << 4;
  const int n0 = ((blockIdx.y << 2) + wave) << 4;
  const int rA = m0 + (lane & 15);
  const int tA = rA >> 6;  // / B
  const int bA = rA & 63;  // % B
  const unsigned short* arow = xb + ((size_t)bA * S + tA) * I;
  v8f acc = {};
  #pragma unroll 4
  for (int kk = 0; kk < I; kk += 32) {
    __builtin_prefetch(arow + kk + 128, 0, 0);  // next A chunks (speculative)
    v16bf a = load_a_frag(arow, kk);
    v16bf b = load_b_frag(Wib, n0, I, kk);
    acc = WMMA_BF16(a, b, acc);
  }
  const int col = n0 + (lane & 15);
  const float bias = bi[col] + bh[col];
  const int mhi = (lane >> 4) << 3;
  #pragma unroll
  for (int r = 0; r < 8; ++r) {
    const int mrow = m0 + r + mhi;
    Pb[(size_t)mrow * H + col] = f32_to_bf16_rne(acc[r] + bias);
  }
}

// ---------------- one recurrence half-step: out = tanh(hin@Wh^T + add) ------
// add = P slice (bf16, per-element, bi+bh folded in) OR bias vector (f32).
// grid (B/16, H/128); 128 threads = 4 waves, 2 N-tiles per wave (A reuse).
__global__ void k_step(const unsigned short* __restrict__ hin,
                       const unsigned short* __restrict__ Whb,
                       const unsigned short* __restrict__ Padd,  // nullable
                       const float* __restrict__ bias,           // nullable
                       unsigned short* __restrict__ outp,
                       size_t outRowStride) {
  const int lane = threadIdx.x & 31;
  const int wave = threadIdx.x >> 5;
  const int m0 = blockIdx.x << 4;
  const int n0 = ((blockIdx.y << 2) + wave) << 5;  // 32 cols per wave
  const unsigned short* arow = hin + (size_t)(m0 + (lane & 15)) * H;
  // P tile is streamed exactly once -> prefetch it while the GEMM runs
  if (Padd) __builtin_prefetch(Padd + (size_t)(m0 + (lane & 15)) * H + n0, 0, 0);
  v8f acc[2] = {{}, {}};
  #pragma unroll 4
  for (int kk = 0; kk < H; kk += 32) {
    v16bf a  = load_a_frag(arow, kk);
    v16bf b0 = load_b_frag(Whb, n0, H, kk);
    v16bf b1 = load_b_frag(Whb, n0 + 16, H, kk);
    acc[0] = WMMA_BF16(a, b0, acc[0]);
    acc[1] = WMMA_BF16(a, b1, acc[1]);
  }
  const int mhi = (lane >> 4) << 3;
  #pragma unroll
  for (int ti = 0; ti < 2; ++ti) {
    const int col = n0 + (ti << 4) + (lane & 15);
    const float cb = bias ? bias[col] : 0.0f;
    #pragma unroll
    for (int r = 0; r < 8; ++r) {
      const int mrow = m0 + r + mhi;
      float v = acc[ti][r] + cb;
      if (Padd) v += bf16_to_f32(Padd[(size_t)mrow * H + col]);
      outp[(size_t)mrow * outRowStride + col] = f32_to_bf16_rne(tanhf(v));
    }
  }
}

// ---------------- out = softmax(z@W3^T + b3) fused ----------------
// 16 rows/block over full N=512; 256 threads = 8 waves x 4 tiles.
// The 16x32 A-tile (z rows) is shared by all 8 waves: stage it via async
// global->LDS copies in a 3-deep ring (ASYNCcnt pipelined), read as ds_load.
__global__ void k_out_softmax(const unsigned short* __restrict__ zb,
                              const unsigned short* __restrict__ W3b,
                              const float* __restrict__ b3,
                              float* __restrict__ outp) {
  __shared__ float ls[16][O];  // 32 KB logits staging for softmax
#if HAVE_ASYNC_COPY
  __shared__ __align__(16) unsigned short abuf[3][16 * 32];  // 3 x 1 KB ring
#endif
  const int lane = threadIdx.x & 31;
  const int wave = threadIdx.x >> 5;
  const int m0 = blockIdx.x << 4;  // flat row = b*S + s
  const int n0 = wave << 6;        // 64 cols per wave
  v8f acc[4] = {{}, {}, {}, {}};
  constexpr int NK = H / 32;

#if HAVE_ASYNC_COPY
  // 64 chunks of 16B cover the 16x32 tile; wave w issues chunks w*8..w*8+7
  // on lanes 0..7 (one async instruction per wave per stage).
  const int c = (wave << 3) + lane;       // chunk id (valid for lane<8)
  const int crow = c >> 2;                // tile row 0..15
  const int ccol = (c & 3) << 3;          // col8 group: 0,8,16,24
  const unsigned short* gsrc = zb + (size_t)(m0 + crow) * H + ccol;

  auto issue = [&](int buf, int kk) {
    if (lane < 8) {
      unsigned short* gp = const_cast<unsigned short*>(gsrc + kk);
      unsigned short* lp = &abuf[buf][(crow << 5) + ccol];
      __builtin_amdgcn_global_load_async_to_lds_b128(
          (AS1 v4i_*)gp, (AS3 v4i_*)lp, 0, 0);
    }
  };
  auto compute = [&](int i) {
    const int kk = i << 5;
    v16bf a = load_a_frag(&abuf[i % 3][(lane & 15) << 5], 0);  // ds_load_b128 x2
    #pragma unroll
    for (int ti = 0; ti < 4; ++ti) {
      v16bf b = load_b_frag(W3b, n0 + (ti << 4), H, kk);
      acc[ti] = WMMA_BF16(a, b, acc[ti]);
    }
  };

  issue(0, 0);
  for (int i = 0; i < NK - 1; ++i) {
    issue((i + 1) % 3, (i << 5) + 32);  // write buf i+1 (last read at i-2)
    WAIT_ASYNC(1);                       // stage i complete, i+1 in flight
    __syncthreads();
    compute(i);
  }
  WAIT_ASYNC(0);
  __syncthreads();
  compute(NK - 1);
#else
  const unsigned short* arow = zb + (size_t)(m0 + (lane & 15)) * H;
  for (int kk = 0; kk < H; kk += 32) {
    v16bf a = load_a_frag(arow, kk);
    #pragma unroll
    for (int ti = 0; ti < 4; ++ti) {
      v16bf b = load_b_frag(W3b, n0 + (ti << 4), H, kk);
      acc[ti] = WMMA_BF16(a, b, acc[ti]);
    }
  }
#endif

  const int mhi = (lane >> 4) << 3;
  #pragma unroll
  for (int ti = 0; ti < 4; ++ti) {
    const int col = n0 + (ti << 4) + (lane & 15);
    const float cb = b3[col];
    #pragma unroll
    for (int r = 0; r < 8; ++r) ls[r + mhi][col] = acc[ti][r] + cb;
  }
  __syncthreads();
  // row softmax: 16 threads/row (half-wave), shuffle reductions
  const int row = threadIdx.x >> 4;
  const int j   = threadIdx.x & 15;
  float mx = -3.402823466e38f;
  #pragma unroll
  for (int k = 0; k < O / 16; ++k) mx = fmaxf(mx, ls[row][j + (k << 4)]);
  for (int off = 8; off > 0; off >>= 1) mx = fmaxf(mx, __shfl_xor(mx, off, 16));
  float sum = 0.f;
  #pragma unroll
  for (int k = 0; k < O / 16; ++k) {
    float e = __expf(ls[row][j + (k << 4)] - mx);
    ls[row][j + (k << 4)] = e;
    sum += e;
  }
  for (int off = 8; off > 0; off >>= 1) sum += __shfl_xor(sum, off, 16);
  const float inv = 1.0f / sum;
  float* orow = outp + (size_t)(m0 + row) * O;
  #pragma unroll
  for (int k = 0; k < O / 16; ++k) orow[j + (k << 4)] = ls[row][j + (k << 4)] * inv;
}

static void launch_cvt(const float* s, unsigned short* d, long n, hipStream_t stream) {
  int blocks = (int)((n + 255) / 256);
  if (blocks > 4096) blocks = 4096;
  k_f32_to_bf16<<<blocks, 256, 0, stream>>>(s, d, n);
}

extern "C" void kernel_launch(void* const* d_in, const int* in_sizes, int n_in,
                              void* d_out, int out_size, void* d_ws, size_t ws_size,
                              hipStream_t stream) {
  (void)in_sizes; (void)n_in; (void)out_size; (void)ws_size;
  const float* x  = (const float*)d_in[0];
  const float* h0 = (const float*)d_in[1];
  const float* Wi = (const float*)d_in[2];
  const float* bi = (const float*)d_in[3];
  const float* Wh = (const float*)d_in[4];
  const float* bh = (const float*)d_in[5];
  const float* W3 = (const float*)d_in[6];
  const float* b3 = (const float*)d_in[7];
  float* out = (float*)d_out;  // (B,S,O) f32, then ht (B,H) f32

  char* ws = (char*)d_ws;
  size_t off = 0;
  auto alloc = [&](size_t bytes) -> char* {
    char* p = ws + off;
    off = (off + bytes + 255) & ~(size_t)255;
    return p;
  };
  unsigned short* xb  = (unsigned short*)alloc((size_t)B * S * I * 2);
  unsigned short* Wib = (unsigned short*)alloc((size_t)H * I * 2);
  unsigned short* Whb = (unsigned short*)alloc((size_t)H * H * 2);
  unsigned short* W3b = (unsigned short*)alloc((size_t)O * H * 2);
  unsigned short* Pb  = (unsigned short*)alloc((size_t)(S - 1) * B * H * 2);
  unsigned short* zb  = (unsigned short*)alloc((size_t)B * S * H * 2);
  unsigned short* hb0 = (unsigned short*)alloc((size_t)B * H * 2);
  unsigned short* hb1 = (unsigned short*)alloc((size_t)B * H * 2);
  unsigned short* hbuf[2] = {hb0, hb1};

  // bf16 conversions (parallel, independent)
  launch_cvt(x,  xb,  (long)B * S * I, stream);
  launch_cvt(Wi, Wib, (long)H * I, stream);
  launch_cvt(Wh, Whb, (long)H * H, stream);
  launch_cvt(W3, W3b, (long)O * H, stream);
  launch_cvt(h0, hb0, (long)B * H, stream);
  k_zero_z0<<<(B * H) / 256, 256, 0, stream>>>(zb);

  // hoisted input projection: P[t*B+b, :] = x[b,t,:]@Wi^T + bi + bh
  k_pgemm<<<dim3((S - 1) * B / 16, H / 64), 128, 0, stream>>>(xb, Wib, bi, bh, Pb);

  // sequential recurrence: two dependent WMMA GEMMs per step
  for (int t = 0; t < S - 1; ++t) {
    const unsigned short* hin = hbuf[t & 1];
    unsigned short* hout = hbuf[(t + 1) & 1];
    // h' = tanh(P[t] + h@Wh^T)
    k_step<<<dim3(B / 16, H / 128), 128, 0, stream>>>(
        hin, Whb, Pb + (size_t)t * B * H, nullptr, hout, (size_t)H);
    // z[:, t+1, :] = tanh(h'@Wh^T + bh)
    k_step<<<dim3(B / 16, H / 128), 128, 0, stream>>>(
        hout, Whb, nullptr, bh, zb + (size_t)(t + 1) * H, (size_t)S * H);
  }

  // out = softmax(z@W3^T + b3)
  k_out_softmax<<<(B * S) / 16, 256, 0, stream>>>(zb, W3b, b3, out);

  // final hidden state -> f32 tail of d_out
  k_bf16_to_f32<<<(B * H) / 256, 256, 0, stream>>>(hbuf[(S - 1) & 1],
                                                   out + (size_t)B * S * O,
                                                   (long)B * H);
}
